// GastGC_24438363914636
// MI455X (gfx1250) — matrix operand order repaired and checked by reading, not compiled, not run on hardware
//
#include <hip/hip_runtime.h>
#include <hip/hip_bf16.h>

// ---------------- problem constants ----------------
#define FEAT_IN 64
#define EMB 256
#define KSORT 30
#define BATCH 48
#define NG 11
#define GR (BATCH * NG)        // 528 graphs
#define NPG 192                // nodes per graph
#define EPG 384                // edges per graph
#define NN (GR * NPG)          // 101376 nodes
#define EE (GR * EPG)          // 202752 edges
#define DCH 769                // 3*EMB+1
#define KP5 800                // DCH padded to mult of 32
#define M5 (GR * KSORT)        // 15840
#define M6 (GR * NG)           // 5808  (11 conv-6 positions per graph)
#define K6 640                 // 128 ch * 5 taps
#define K7 30976               // NG * 2816
#define DENSE 2816

typedef __attribute__((ext_vector_type(16))) __bf16 v16bf;
typedef __attribute__((ext_vector_type(8)))  float  f32x8;

// =====================================================================
// bf16 WMMA GEMM: C[M,Nn] = A[M,Kp](bf16 rowmajor) * Bpacked
// B packed fragment layout (k-major so one k-step is contiguous):
//   Bp[ ((kt*NTtot + nt)*32 + lane)*16 + e ]
//   with lane = n%16 + 16*((k%32)/16), e = k%16.
// Each wave computes a 16x(16*NTILE) tile; B fragments for one kt sit at
// compile-time immediate offsets from one pointer bumped per iteration.
// fuse!=0 -> +bias, ReLU.
// =====================================================================
template <int NTILE>
__global__ void gemm_bf16_wmma(const __bf16* __restrict__ A,
                               const __bf16* __restrict__ Bp,
                               float* __restrict__ C,
                               const float* __restrict__ bias,
                               int M, int Kp, int Nn, int fuse) {
    const int lane  = threadIdx.x & 31;
    const int wave  = threadIdx.x >> 5;
    const int mtile = blockIdx.x * 8 + wave;
    if (mtile * 16 >= M) return;
    const int KT    = Kp >> 5;
    const int NTtot = Nn >> 4;
    const int nt0   = blockIdx.y * NTILE;
    const int mrow  = lane & 15;
    const int khalf = lane >> 4;             // 0: K 0-15, 1: K 16-31

    const __bf16* aptr = A + (size_t)(mtile * 16 + mrow) * Kp + khalf * 16;
    const __bf16* bptr = Bp + ((size_t)nt0 * 32 + lane) * 16;
    const size_t  bstep = (size_t)NTtot * 512;          // elements per kt

    f32x8 acc[NTILE];
#pragma unroll
    for (int t = 0; t < NTILE; ++t)
#pragma unroll
        for (int i = 0; i < 8; ++i) acc[t][i] = 0.0f;

    for (int kt = 0; kt < KT; ++kt) {
        union { uint4 u[2]; v16bf v; } af;
        const uint4* ap = (const uint4*)aptr;
        af.u[0] = ap[0];     // k offsets +0..7   (elems 0-7)
        af.u[1] = ap[1];     // k offsets +8..15  (elems 8-15)
#pragma unroll
        for (int t = 0; t < NTILE; ++t) {
            union { uint4 u[2]; v16bf v; } bf;
            const uint4* bp = (const uint4*)(bptr + t * 512);  // +t*1024 bytes (imm)
            bf.u[0] = bp[0];
            bf.u[1] = bp[1];
            acc[t] = __builtin_amdgcn_wmma_f32_16x16x32_bf16(
                false, af.v, false, bf.v, (short)0, acc[t], false, false);
        }
        aptr += 32;          // 64 bytes
        bptr += bstep;       // NTtot*1024 bytes
    }
    const int crow0 = mtile * 16 + khalf * 8;
#pragma unroll
    for (int t = 0; t < NTILE; ++t) {
        const int col = (nt0 + t) * 16 + mrow;
        float bv = fuse ? bias[col] : 0.0f;
#pragma unroll
        for (int i = 0; i < 8; ++i) {
            float v = acc[t][i] + bv;
            if (fuse) v = fmaxf(v, 0.0f);
            C[(size_t)(crow0 + i) * Nn + col] = v;
        }
    }
}

// ---------------- weight pre-swizzle into B-fragment layout ----------------
// Output layout: [kt][nt][lane][16]  (k-major, see GEMM above)
// mode 0: src is f32 [Ksrc, Nn] row-major      (W1,W2,W3,Wc1)
// mode 1: src is w5 f32 [Nn, Ksrc]  (transpose)
// mode 2: src is w6 f32 [Nn, 128, 5]; k -> (ic=k/5, t=k%5)
__global__ void pack_b_kernel(const float* __restrict__ src,
                              __bf16* __restrict__ dst,
                              int Ksrc, int Kp, int Nn, int mode) {
    size_t tid = (size_t)blockIdx.x * blockDim.x + threadIdx.x;
    size_t total = (size_t)Kp * Nn;
    if (tid >= total) return;
    int e    = tid & 15;
    int lane = (tid >> 4) & 31;
    size_t rest = tid >> 9;
    int NTtot = Nn >> 4;
    int nt = (int)(rest % NTtot);
    int kt = (int)(rest / NTtot);
    int k  = kt * 32 + (lane >> 4) * 16 + e;
    int n  = nt * 16 + (lane & 15);
    float v = 0.0f;
    if (k < Ksrc) {
        if (mode == 0)      v = src[(size_t)k * Nn + n];
        else if (mode == 1) v = src[(size_t)n * Ksrc + k];
        else { int ic = k / 5, t = k - ic * 5; v = src[((size_t)n * 128 + ic) * 5 + t]; }
    }
    dst[tid] = (__bf16)v;
}

// ---------------- misc elementwise ----------------
__global__ void cast_f32_bf16(const float* __restrict__ s, __bf16* __restrict__ d, size_t n) {
    size_t i = (size_t)blockIdx.x * blockDim.x + threadIdx.x;
    if (i < n) d[i] = (__bf16)s[i];
}
__global__ void deg_init(float* deg, int n) {
    int i = blockIdx.x * blockDim.x + threadIdx.x;
    if (i < n) deg[i] = 1.0f;
}
__global__ void deg_accum(const int* __restrict__ dst, float* deg, int e) {
    int i = blockIdx.x * blockDim.x + threadIdx.x;
    if (i < e) atomicAdd(&deg[dst[i]], 1.0f);
}
__global__ void deg_to_dis(float* d, int n) {
    int i = blockIdx.x * blockDim.x + threadIdx.x;
    if (i < n) d[i] = rsqrtf(d[i]);
}

// =====================================================================
// GCN aggregation fully on-chip: one block per (graph, 64-channel slab).
// agg lives in LDS (192*64 f32 = 48KB) and is fed with ds_add_f32.
// Fused epilogue: tanh(agg + h*dis^2 + b) -> bf16 next-layer activation.
// =====================================================================
#define CSLAB 64
__global__ void gcn_agg_tanh(const float* __restrict__ H,
                             const int* __restrict__ src,
                             const int* __restrict__ dst,
                             const float* __restrict__ dis,
                             const float* __restrict__ bias,
                             __bf16* __restrict__ xout) {
    __shared__ float sAgg[NPG * CSLAB];
    __shared__ int   sS[EPG];
    __shared__ short sD[EPG];
    __shared__ float sC[EPG];
    const int g     = blockIdx.x;
    const int cbase = blockIdx.y * CSLAB;
    const int nbase = g * NPG;
    for (int i = threadIdx.x; i < NPG * CSLAB; i += 256) sAgg[i] = 0.0f;
    for (int e = threadIdx.x; e < EPG; e += 256) {
        int s = src[g * EPG + e], d = dst[g * EPG + e];
        sS[e] = s; sD[e] = (short)(d - nbase); sC[e] = dis[s] * dis[d];
    }
    __syncthreads();
    // 384 edges x 16 channel-quads
    for (int it = threadIdx.x; it < EPG * 16; it += 256) {
        int e = it >> 4, cg = it & 15;
        int s = sS[e];
        float coef = sC[e];
        const float4 hv = *(const float4*)(H + (size_t)s * EMB + cbase + cg * 4);
        int ld = (int)sD[e] * CSLAB + cg * 4;
        atomicAdd(&sAgg[ld + 0], hv.x * coef);
        atomicAdd(&sAgg[ld + 1], hv.y * coef);
        atomicAdd(&sAgg[ld + 2], hv.z * coef);
        atomicAdd(&sAgg[ld + 3], hv.w * coef);
    }
    __syncthreads();
    for (int it = threadIdx.x; it < NPG * CSLAB; it += 256) {
        int ln = it >> 6, c = it & (CSLAB - 1);
        int n = nbase + ln;
        float dd = dis[n];
        float v = sAgg[it] + H[(size_t)n * EMB + cbase + c] * (dd * dd) + bias[cbase + c];
        xout[(size_t)n * EMB + cbase + c] = (__bf16)tanhf(v);
    }
}

// ---- GCN layer 4 (scalar channel), fully per-graph in LDS ----
__global__ void gcn4_kernel(const __bf16* __restrict__ x3b,
                            const int* __restrict__ src,
                            const int* __restrict__ dst,
                            const float* __restrict__ dis,
                            const float* __restrict__ W4,
                            const float* __restrict__ b4,
                            float* __restrict__ x4) {
    __shared__ float sH[NPG];
    __shared__ float sA[NPG];
    const int g = blockIdx.x, nbase = g * NPG, t = threadIdx.x;
    if (t < NPG) {
        const __bf16* row = x3b + (size_t)(nbase + t) * EMB;
        float acc = 0.0f;
        for (int i = 0; i < EMB; ++i) acc += (float)row[i] * W4[i];
        sH[t] = acc; sA[t] = 0.0f;
    }
    __syncthreads();
    for (int e = t; e < EPG; e += 256) {
        int s = src[g * EPG + e], d = dst[g * EPG + e];
        atomicAdd(&sA[d - nbase], sH[s - nbase] * dis[s] * dis[d]);
    }
    __syncthreads();
    if (t < NPG) {
        float dd = dis[nbase + t];
        x4[nbase + t] = tanhf(sA[t] + sH[t] * dd * dd + b4[0]);
    }
}

// ---- per-graph top-30 by x4 (descending, stable) : bitonic sort 256 ----
__global__ void topk_kernel(const float* __restrict__ x4, int* __restrict__ idxK) {
    __shared__ float sK[256];
    __shared__ int   sI[256];
    const int g = blockIdx.x, t = threadIdx.x;
    sK[t] = (t < NPG) ? x4[g * NPG + t] : -3.4e38f;
    sI[t] = t;
    __syncthreads();
    for (int k = 2; k <= 256; k <<= 1) {
        for (int j = k >> 1; j > 0; j >>= 1) {
            int ixj = t ^ j;
            if (ixj > t) {
                bool desc = ((t & k) == 0);
                float ka = sK[t], kb = sK[ixj];
                int ia = sI[t], ib = sI[ixj];
                bool aFirst = (ka > kb) || (ka == kb && ia < ib);
                if (desc ? !aFirst : aFirst) {
                    sK[t] = kb; sK[ixj] = ka; sI[t] = ib; sI[ixj] = ia;
                }
            }
            __syncthreads();
        }
    }
    if (t < KSORT) idxK[g * KSORT + t] = sI[t];
}

// ---- build conv5 im2col A [M5, KP5] bf16 (concat x1..x4 of top-K nodes) ----
__global__ void build_a5(const __bf16* __restrict__ x1b, const __bf16* __restrict__ x2b,
                         const __bf16* __restrict__ x3b, const float* __restrict__ x4,
                         const int* __restrict__ idxK, __bf16* __restrict__ A5) {
    size_t tid = (size_t)blockIdx.x * blockDim.x + threadIdx.x;
    if (tid >= (size_t)M5 * KP5) return;
    int row = (int)(tid / KP5), col = (int)(tid % KP5);
    int g = row / KSORT, kk = row % KSORT;
    int n = g * NPG + idxK[g * KSORT + kk];
    __bf16 v = (__bf16)0.0f;
    if (col < 256)       v = x1b[(size_t)n * EMB + col];
    else if (col < 512)  v = x2b[(size_t)n * EMB + col - 256];
    else if (col < 768)  v = x3b[(size_t)n * EMB + col - 512];
    else if (col == 768) v = (__bf16)x4[n];
    A5[tid] = v;
}

// ---- maxpool(2,2) fused with conv6 im2col: A6[(g,p), ic*5+t] ----
__global__ void build_a6(const float* __restrict__ C5, __bf16* __restrict__ A6) {
    size_t tid = (size_t)blockIdx.x * blockDim.x + threadIdx.x;
    if (tid >= (size_t)M6 * K6) return;
    int row = (int)(tid / K6), col = (int)(tid % K6);
    int g = row / NG, p = row % NG;
    int ic = col / 5, t = col % 5;
    int kk = p + t;                                   // pooled position 0..14
    float a = C5[((size_t)g * KSORT + 2 * kk) * 128 + ic];
    float b = C5[((size_t)g * KSORT + 2 * kk + 1) * 128 + ic];
    A6[tid] = (__bf16)fmaxf(a, b);
}

// ---- flatten conv6 output into dense-layer A [48, 30976] bf16 ----
__global__ void build_a7(const float* __restrict__ C6, __bf16* __restrict__ A7) {
    size_t tid = (size_t)blockIdx.x * blockDim.x + threadIdx.x;
    if (tid >= (size_t)BATCH * K7) return;
    int b = (int)(tid / K7), col = (int)(tid % K7);
    int j = col / DENSE, r = col % DENSE;
    int oc = r / NG, pos = r % NG;
    A7[tid] = (__bf16)C6[((size_t)b * (NG * NG) + j * NG + pos) * EMB + oc];
}

// ---- final tiny dense: logits = H7 @ Wc2 + bc2 ----
__global__ void dense2_kernel(const float* __restrict__ H7, const float* __restrict__ Wc2,
                              const float* __restrict__ bc2, float* __restrict__ out) {
    int tid = threadIdx.x;
    if (tid >= BATCH * 10) return;
    int b = tid / 10, o = tid % 10;
    float acc = bc2[o];
    for (int i = 0; i < EMB; ++i) acc += H7[b * EMB + i] * Wc2[i * 10 + o];
    out[b * 10 + o] = acc;
}

// =====================================================================
extern "C" void kernel_launch(void* const* d_in, const int* in_sizes, int n_in,
                              void* d_out, int out_size, void* d_ws, size_t ws_size,
                              hipStream_t stream) {
    const float* x   = (const float*)d_in[0];
    const int*   src = (const int*)d_in[1];
    const int*   dst = (const int*)d_in[2];
    const float* W1  = (const float*)d_in[3];  const float* b1 = (const float*)d_in[4];
    const float* W2  = (const float*)d_in[5];  const float* b2 = (const float*)d_in[6];
    const float* W3  = (const float*)d_in[7];  const float* b3 = (const float*)d_in[8];
    const float* W4  = (const float*)d_in[9];  const float* b4 = (const float*)d_in[10];
    const float* w5  = (const float*)d_in[11]; const float* bc5 = (const float*)d_in[12];
    const float* w6  = (const float*)d_in[13]; const float* bc6 = (const float*)d_in[14];
    const float* Wc1 = (const float*)d_in[15]; const float* bc1 = (const float*)d_in[16];
    const float* Wc2 = (const float*)d_in[17]; const float* bc2 = (const float*)d_in[18];
    float* out = (float*)d_out;

    // ---- workspace carve-up ----
    char* ws = (char*)d_ws;
    size_t off = 0;
    auto alloc = [&](size_t bytes) { void* p = ws + off; off = (off + bytes + 255) & ~(size_t)255; return p; };
    __bf16* x0b = (__bf16*)alloc((size_t)NN * FEAT_IN * 2);
    __bf16* W1p = (__bf16*)alloc((size_t)FEAT_IN * EMB * 2);
    __bf16* W2p = (__bf16*)alloc((size_t)EMB * EMB * 2);
    __bf16* W3p = (__bf16*)alloc((size_t)EMB * EMB * 2);
    __bf16* B5p = (__bf16*)alloc((size_t)KP5 * 128 * 2);
    __bf16* B6p = (__bf16*)alloc((size_t)K6 * EMB * 2);
    __bf16* B7p = (__bf16*)alloc((size_t)K7 * EMB * 2);
    float*  H   = (float*)alloc((size_t)NN * EMB * 4);
    __bf16* x1b = (__bf16*)alloc((size_t)NN * EMB * 2);
    __bf16* x2b = (__bf16*)alloc((size_t)NN * EMB * 2);
    __bf16* x3b = (__bf16*)alloc((size_t)NN * EMB * 2);
    float*  x4  = (float*)alloc((size_t)NN * 4);
    float*  dis = (float*)alloc((size_t)NN * 4);
    int*    idxK= (int*)alloc((size_t)GR * KSORT * 4);
    __bf16* A5  = (__bf16*)alloc((size_t)M5 * KP5 * 2);
    float*  C5  = (float*)alloc((size_t)M5 * 128 * 4);
    __bf16* A6  = (__bf16*)alloc((size_t)M6 * K6 * 2);
    float*  C6  = (float*)alloc((size_t)M6 * EMB * 4);
    __bf16* A7  = (__bf16*)alloc((size_t)BATCH * K7 * 2);
    float*  H7  = (float*)alloc((size_t)BATCH * EMB * 4);

    auto cdiv = [](size_t a, size_t b) { return (unsigned)((a + b - 1) / b); };

    // ---- casts & weight pre-swizzles ----
    cast_f32_bf16<<<cdiv((size_t)NN * FEAT_IN, 256), 256, 0, stream>>>(x, x0b, (size_t)NN * FEAT_IN);
    pack_b_kernel<<<cdiv((size_t)FEAT_IN * EMB, 256), 256, 0, stream>>>(W1, W1p, FEAT_IN, FEAT_IN, EMB, 0);
    pack_b_kernel<<<cdiv((size_t)EMB * EMB, 256), 256, 0, stream>>>(W2, W2p, EMB, EMB, EMB, 0);
    pack_b_kernel<<<cdiv((size_t)EMB * EMB, 256), 256, 0, stream>>>(W3, W3p, EMB, EMB, EMB, 0);
    pack_b_kernel<<<cdiv((size_t)KP5 * 128, 256), 256, 0, stream>>>(w5, B5p, DCH, KP5, 128, 1);
    pack_b_kernel<<<cdiv((size_t)K6 * EMB, 256), 256, 0, stream>>>(w6, B6p, K6, K6, EMB, 2);
    pack_b_kernel<<<cdiv((size_t)K7 * EMB, 256), 256, 0, stream>>>(Wc1, B7p, K7, K7, EMB, 0);

    // ---- degree / symmetric norm ----
    deg_init<<<cdiv(NN, 256), 256, 0, stream>>>(dis, NN);
    deg_accum<<<cdiv(EE, 256), 256, 0, stream>>>(dst, dis, EE);
    deg_to_dis<<<cdiv(NN, 256), 256, 0, stream>>>(dis, NN);

    // ---- 3 GCN layers: WMMA GEMM (16x128 wave tiles) + on-chip aggregation ----
    dim3 gemmBlk(256);
    dim3 aggGrid(GR, EMB / CSLAB);
    gemm_bf16_wmma<8><<<dim3(NN / 16 / 8, EMB / 128), gemmBlk, 0, stream>>>(x0b, W1p, H, nullptr, NN, FEAT_IN, EMB, 0);
    gcn_agg_tanh<<<aggGrid, 256, 0, stream>>>(H, src, dst, dis, b1, x1b);
    gemm_bf16_wmma<8><<<dim3(NN / 16 / 8, EMB / 128), gemmBlk, 0, stream>>>(x1b, W2p, H, nullptr, NN, EMB, EMB, 0);
    gcn_agg_tanh<<<aggGrid, 256, 0, stream>>>(H, src, dst, dis, b2, x2b);
    gemm_bf16_wmma<8><<<dim3(NN / 16 / 8, EMB / 128), gemmBlk, 0, stream>>>(x2b, W3p, H, nullptr, NN, EMB, EMB, 0);
    gcn_agg_tanh<<<aggGrid, 256, 0, stream>>>(H, src, dst, dis, b3, x3b);
    gcn4_kernel<<<GR, 256, 0, stream>>>(x3b, src, dst, dis, W4, b4, x4);

    // ---- sort-pool + conv5 GEMM ----
    topk_kernel<<<GR, 256, 0, stream>>>(x4, idxK);
    build_a5<<<cdiv((size_t)M5 * KP5, 256), 256, 0, stream>>>(x1b, x2b, x3b, x4, idxK, A5);
    gemm_bf16_wmma<8><<<dim3(cdiv(M5 / 16, 8), 128 / 128), gemmBlk, 0, stream>>>(A5, B5p, C5, bc5, M5, KP5, 128, 1);

    // ---- maxpool + conv6 GEMM ----
    build_a6<<<cdiv((size_t)M6 * K6, 256), 256, 0, stream>>>(C5, A6);
    gemm_bf16_wmma<8><<<dim3(cdiv(M6 / 16, 8), EMB / 128), gemmBlk, 0, stream>>>(A6, B6p, C6, bc6, M6, K6, EMB, 1);

    // ---- dense layers ----
    build_a7<<<cdiv((size_t)BATCH * K7, 256), 256, 0, stream>>>(C6, A7);
    gemm_bf16_wmma<4><<<dim3(1, EMB / 64), gemmBlk, 0, stream>>>(A7, B7p, H7, bc1, BATCH, K7, EMB, 1);
    dense2_kernel<<<1, 512, 0, stream>>>(H7, Wc2, bc2, out);

    (void)in_sizes; (void)n_in; (void)out_size; (void)ws_size;
}
